// MEGNETModel_89395449299362
// MI455X (gfx1250) — compile-verified, matrix operand order checked by reading, not run on hardware
//
#include <hip/hip_runtime.h>
#include <hip/hip_bf16.h>
#include <math.h>

#define DEVINL __device__ __forceinline__

typedef __attribute__((ext_vector_type(16))) _Float16 v16h;
typedef __attribute__((ext_vector_type(8)))  float    v8f;

static constexpr int Nn = 100000;   // nodes
static constexpr int Ee = 800000;   // edges
static constexpr int Bb = 1024;     // graphs

// ---------------------------------------------------------------------------
// Fragment-slot mapping for CDNA5 WMMA 16x16x32 f16 operands.
// A 16-bit operand fragment (v16h, per lane): element e, lane = grp*16 + row,
// covers K index  kk = (p&3)*2 + (p>>2)*16 + grp*8 + j  with p=e>>1, j=e&1.
// frag_slot inverts this: given (kk, row) -> halfword slot within a 512-half
// fragment tile laid out [lane][16 contiguous halves], so that a fragment
// load is one 32-byte contiguous LDS read (2x ds_load_b128).
// ---------------------------------------------------------------------------
DEVINL int frag_slot(int kk, int rowIdx) {
  const int j   = kk & 1;
  const int grp = (kk >> 3) & 1;
  const int pl  = (kk >> 1) & 3;
  const int ph  = kk >> 4;
  const int e   = (((ph << 2) | pl) << 1) | j;
  const int lane = grp * 16 + rowIdx;
  return lane * 16 + e;
}

DEVINL v16h frag_load(const _Float16* tileBase, int lane) {
  return *(const v16h*)(tileBase + lane * 16);
}

// ---------------------------------------------------------------------------
// Input fetch modes for the fused MLP kernel
//   0: direct [M, DIN] f32
//   1: Gaussian expansion of edge_attr (DIN=100)
//   2: edge concat [v_d[row], v_d[col], e_d, u_d[batch[row]]] (DIN=128)
//   3: node/state concat [vd[m], agg[m], ud[bat?bat[m]:m]]    (DIN=96)
// ---------------------------------------------------------------------------
template <int MODE, int DIN>
DEVINL float fetch_in(int gm, int k,
                      const float* in, const float* eattr,
                      const int* row, const int* col, const int* bat,
                      const float* vd, const float* ed, const float* ud,
                      const float* agg) {
  if (MODE == 0) return in[(long)gm * DIN + k];
  if (MODE == 1) {
    const float width = 5.0f / 99.0f;
    float t = (eattr[gm] - (float)k * width) / width;
    return expf(-t * t);
  }
  if (MODE == 2) {
    if (k < 32)  return vd[(long)row[gm] * 32 + k];
    if (k < 64)  return vd[(long)col[gm] * 32 + (k - 32)];
    if (k < 96)  return ed[(long)gm * 32 + (k - 64)];
    return ud[(long)bat[row[gm]] * 32 + (k - 96)];
  }
  // MODE 3
  if (k < 32)  return vd[(long)gm * 32 + k];
  if (k < 64)  return agg[(long)gm * 32 + (k - 32)];
  { int b = bat ? bat[gm] : gm; return ud[(long)b * 32 + (k - 64)]; }
}

// ---------------------------------------------------------------------------
// Fused 2-layer MLP: relu(relu(A @ W1 + b1) @ W2 + b2)
// DIN -> 64 -> 32.  128 threads = 4 waves, 16 rows per wave, 64 rows/block.
// All LDS operands stored pre-swizzled into WMMA fragment layout.
// ---------------------------------------------------------------------------
template <int DIN, int KPAD, int MODE>
__global__ __launch_bounds__(128)
void mlp_wmma_kernel(const float* __restrict__ in, const float* __restrict__ eattr,
                     const int* __restrict__ row, const int* __restrict__ col,
                     const int* __restrict__ bat,
                     const float* __restrict__ vd, const float* __restrict__ ed,
                     const float* __restrict__ ud, const float* __restrict__ agg,
                     const float* __restrict__ W1, const float* __restrict__ b1,
                     const float* __restrict__ W2, const float* __restrict__ b2,
                     float* __restrict__ out, int M) {
  constexpr int KT1 = KPAD / 32;   // layer-1 K tiles
  __shared__ __attribute__((aligned(32))) _Float16 sW1f[4 * KT1 * 512]; // [nt][kt][512]
  __shared__ __attribute__((aligned(32))) _Float16 sW2f[2 * 2 * 512];   // [nt][kt][512]
  __shared__ __attribute__((aligned(32))) _Float16 sAf[4 * KT1 * 512];  // [wave][kt][512]
  __shared__ __attribute__((aligned(32))) _Float16 sHf[4 * 2 * 512];    // [wave][kt][512]
  __shared__ float sb1[64];
  __shared__ float sb2[32];

  const int tid  = threadIdx.x;
  const int lane = tid & 31;
  const int w    = tid >> 5;
  const int rowBase = blockIdx.x * 64 + w * 16;

  // ---- stage weights into fragment layout (f32 -> f16, zero-pad K >= DIN)
  for (int i = tid; i < KPAD * 64; i += 128) {
    int k = i >> 6, n = i & 63;
    float v = (k < DIN) ? W1[k * 64 + n] : 0.0f;
    sW1f[((n >> 4) * KT1 + (k >> 5)) * 512 + frag_slot(k & 31, n & 15)] = (_Float16)v;
  }
  for (int i = tid; i < 64 * 32; i += 128) {
    int k = i >> 5, n = i & 31;
    sW2f[((n >> 4) * 2 + (k >> 5)) * 512 + frag_slot(k & 31, n & 15)] = (_Float16)W2[i];
  }
  if (tid < 64) sb1[tid] = b1[tid];
  if (tid < 32) sb2[tid] = b2[tid];

  // ---- stage A tile (16 x KPAD) for this wave into fragment layout
  _Float16* Af = sAf + w * (KT1 * 512);
  if (MODE == 0 && lane == 0 && rowBase + 64 < M)
    __builtin_prefetch(in + (long)(rowBase + 64) * DIN, 0, 1);
  for (int i = lane; i < 16 * KPAD; i += 32) {
    int r = i / KPAD, k = i % KPAD;
    int gm = rowBase + r;
    float v = 0.0f;
    if (gm < M && k < DIN)
      v = fetch_in<MODE, DIN>(gm, k, in, eattr, row, col, bat, vd, ed, ud, agg);
    Af[(k >> 5) * 512 + frag_slot(k & 31, r)] = (_Float16)v;
  }
  __syncthreads();

  const int n15 = lane & 15, grp = lane >> 4;
  _Float16* Hf = sHf + w * (2 * 512);

  // ---- layer 1: [16,KPAD] x [KPAD,64], 4 accumulators (one per 16-col tile)
  {
    v8f acc[4] = {{}, {}, {}, {}};
#pragma unroll
    for (int kt = 0; kt < KT1; ++kt) {
      v16h af = frag_load(Af + kt * 512, lane);
#pragma unroll
      for (int nt = 0; nt < 4; ++nt) {
        v16h bf = frag_load(sW1f + (nt * KT1 + kt) * 512, lane);
        acc[nt] = __builtin_amdgcn_wmma_f32_16x16x32_f16(false, af, false, bf,
                                                         (short)0, acc[nt], false, false);
      }
    }
    // bias + relu, write H directly in layer-2 A-fragment layout (k2 = n)
#pragma unroll
    for (int nt = 0; nt < 4; ++nt) {
      const int n = nt * 16 + n15;
      const float bias = sb1[n];
#pragma unroll
      for (int r2 = 0; r2 < 8; ++r2) {
        int m = r2 + grp * 8;
        float v = fmaxf(acc[nt][r2] + bias, 0.0f);
        Hf[(n >> 5) * 512 + frag_slot(n & 31, m)] = (_Float16)v;
      }
    }
  }
  __syncthreads();

  // ---- layer 2: [16,64] x [64,32]
  {
    v8f acc[2] = {{}, {}};
#pragma unroll
    for (int kt = 0; kt < 2; ++kt) {
      v16h af = frag_load(Hf + kt * 512, lane);
#pragma unroll
      for (int nt = 0; nt < 2; ++nt) {
        v16h bf = frag_load(sW2f + (nt * 2 + kt) * 512, lane);
        acc[nt] = __builtin_amdgcn_wmma_f32_16x16x32_f16(false, af, false, bf,
                                                         (short)0, acc[nt], false, false);
      }
    }
#pragma unroll
    for (int nt = 0; nt < 2; ++nt) {
      const int n = nt * 16 + n15;
      const float bias = sb2[n];
#pragma unroll
      for (int r2 = 0; r2 < 8; ++r2) {
        int m  = r2 + grp * 8;
        int gm = rowBase + m;
        if (gm < M) out[(long)gm * 32 + n] = fmaxf(acc[nt][r2] + bias, 0.0f);
      }
    }
  }
}

// ---------------------------------------------------------------------------
// Elementwise / scatter helpers
// ---------------------------------------------------------------------------
__global__ void zero_f(float* p, long n) {
  long i = (long)blockIdx.x * blockDim.x + threadIdx.x;
  if (i < n) p[i] = 0.0f;
}
__global__ void zero_u32(unsigned* p, long n) {
  long i = (long)blockIdx.x * blockDim.x + threadIdx.x;
  if (i < n) p[i] = 0u;
}
__global__ void add_f(float* dst, const float* a, const float* b, long n) {
  long i = (long)blockIdx.x * blockDim.x + threadIdx.x;
  if (i < n) dst[i] = a[i] + b[i];
}
__global__ void copy_f(float* dst, const float* src, long n) {
  long i = (long)blockIdx.x * blockDim.x + threadIdx.x;
  if (i < n) dst[i] = src[i];
}
__global__ void embed_k(const int* x, const float* emb, float* v, long n32) {
  long i = (long)blockIdx.x * blockDim.x + threadIdx.x;
  if (i < n32) { int node = (int)(i >> 5), f = (int)(i & 31); v[i] = emb[(long)x[node] * 32 + f]; }
}
__global__ void ebatch_k(const int* row, const int* batch, int* eb, int E) {
  int e = blockIdx.x * blockDim.x + threadIdx.x;
  if (e < E) eb[e] = batch[row[e]];
}
__global__ void scatter_mean_accum(const float* vals, const int* seg, float* acc,
                                   float* cnt, long n32) {
  long i = (long)blockIdx.x * blockDim.x + threadIdx.x;
  if (i >= n32) return;
  int k = (int)(i >> 5), f = (int)(i & 31);
  int b = seg[k];
  atomicAdd(&acc[(long)b * 32 + f], vals[i]);
  if (f == 0) atomicAdd(&cnt[b], 1.0f);
}
__global__ void div_cnt(float* acc, const float* cnt, long n32) {
  long i = (long)blockIdx.x * blockDim.x + threadIdx.x;
  if (i < n32) acc[i] /= fmaxf(cnt[i >> 5], 1.0f);
}

// ---------------------------------------------------------------------------
// Set2Set
// ---------------------------------------------------------------------------
DEVINL float sigm(float x) { return 1.0f / (1.0f + expf(-x)); }
DEVINL unsigned ord_enc(float x) {
  unsigned b = __float_as_uint(x);
  return (b & 0x80000000u) ? ~b : (b | 0x80000000u);
}
DEVINL float ord_dec(unsigned v) {
  unsigned b = (v & 0x80000000u) ? (v ^ 0x80000000u) : ~v;
  return __uint_as_float(b);
}

__global__ __launch_bounds__(128)
void lstm_step_kernel(const float* __restrict__ Wih, const float* __restrict__ Whh,
                      const float* __restrict__ bih, const float* __restrict__ bhh,
                      const float* __restrict__ qstar, float* h, float* c) {
  __shared__ float gates[128];
  int b = blockIdx.x, g = threadIdx.x;
  float acc = bih[g] + bhh[g];
  for (int j = 0; j < 64; ++j) acc += qstar[(long)b * 64 + j] * Wih[g * 64 + j];
  for (int j = 0; j < 32; ++j) acc += h[(long)b * 32 + j] * Whh[g * 32 + j];
  gates[g] = acc;
  __syncthreads();
  if (g < 32) {
    float ig = sigm(gates[g]);
    float fg = sigm(gates[32 + g]);
    float gg = tanhf(gates[64 + g]);
    float og = sigm(gates[96 + g]);
    float cn = fg * c[(long)b * 32 + g] + ig * gg;
    c[(long)b * 32 + g] = cn;
    h[(long)b * 32 + g] = og * tanhf(cn);
  }
}

__global__ void s2s_logit_kernel(const float* xv, const float* h, const int* seg,
                                 float* logit, unsigned* mord, int K) {
  int k = blockIdx.x * blockDim.x + threadIdx.x;
  if (k >= K) return;
  int b = seg[k];
  float acc = 0.0f;
  for (int f = 0; f < 32; ++f) acc += xv[(long)k * 32 + f] * h[(long)b * 32 + f];
  logit[k] = acc;
  atomicMax(&mord[b], ord_enc(acc));
}
__global__ void s2s_exp_kernel(const float* logit, const int* seg, const unsigned* mord,
                               float* ex, float* s, int K) {
  int k = blockIdx.x * blockDim.x + threadIdx.x;
  if (k >= K) return;
  int b = seg[k];
  unsigned mv = mord[b];
  float m = (mv == 0u) ? 0.0f : ord_dec(mv);
  float v = expf(logit[k] - m);
  ex[k] = v;
  atomicAdd(&s[b], v);
}
__global__ void s2s_r_kernel(const float* xv, const int* seg, const float* ex,
                             const float* s, float* r, long n32) {
  long i = (long)blockIdx.x * blockDim.x + threadIdx.x;
  if (i >= n32) return;
  int k = (int)(i >> 5), f = (int)(i & 31);
  int b = seg[k];
  float a = ex[k] / (s[b] + 1e-16f);
  atomicAdd(&r[(long)b * 32 + f], a * xv[i]);
}
__global__ void s2s_qstar_kernel(const float* h, const float* r, float* qstar, long b32) {
  long i = (long)blockIdx.x * blockDim.x + threadIdx.x;
  if (i >= b32) return;
  int b = (int)(i >> 5), f = (int)(i & 31);
  qstar[(long)b * 64 + f]      = h[i];
  qstar[(long)b * 64 + 32 + f] = r[i];
}

// ---------------------------------------------------------------------------
// Readout head: z=[vs(64), es(64), u(32)] -> 32 -> 16 -> 1
// ---------------------------------------------------------------------------
__global__ void head_kernel(const float* qv, const float* qe, const float* u,
                            const float* W1, const float* b1,
                            const float* W2, const float* b2,
                            const float* Wo, const float* bo, float* outp, int B) {
  int b = blockIdx.x * blockDim.x + threadIdx.x;
  if (b >= B) return;
  float z[160];
  for (int i = 0; i < 64; ++i) z[i] = qv[(long)b * 64 + i];
  for (int i = 0; i < 64; ++i) z[64 + i] = qe[(long)b * 64 + i];
  for (int i = 0; i < 32; ++i) z[128 + i] = u[(long)b * 32 + i];
  float h1[32];
  for (int n = 0; n < 32; ++n) {
    float a = b1[n];
    for (int k = 0; k < 160; ++k) a += z[k] * W1[k * 32 + n];
    h1[n] = fmaxf(a, 0.0f);
  }
  float h2[16];
  for (int n = 0; n < 16; ++n) {
    float a = b2[n];
    for (int k = 0; k < 32; ++k) a += h1[k] * W2[k * 16 + n];
    h2[n] = fmaxf(a, 0.0f);
  }
  float a = bo[0];
  for (int k = 0; k < 16; ++k) a += h2[k] * Wo[k];
  outp[b] = a;
}

// ---------------------------------------------------------------------------
// Host side
// ---------------------------------------------------------------------------
static inline unsigned cdivu(long a, long b) { return (unsigned)((a + b - 1) / b); }

struct S2SParams { const float* Wih; const float* Whh; const float* bih; const float* bhh; };

static void run_set2set(const float* xv, const int* seg, int K, S2SParams p,
                        float* hbuf, float* cbuf, float* qstar,
                        float* logit, float* exbuf, unsigned* mord, float* sbuf,
                        float* rbuf, hipStream_t stream) {
  zero_f<<<cdivu(Bb * 32, 256), 256, 0, stream>>>(hbuf, Bb * 32);
  zero_f<<<cdivu(Bb * 32, 256), 256, 0, stream>>>(cbuf, Bb * 32);
  zero_f<<<cdivu(Bb * 64, 256), 256, 0, stream>>>(qstar, Bb * 64);
  for (int step = 0; step < 3; ++step) {
    lstm_step_kernel<<<Bb, 128, 0, stream>>>(p.Wih, p.Whh, p.bih, p.bhh, qstar, hbuf, cbuf);
    zero_u32<<<cdivu(Bb, 256), 256, 0, stream>>>(mord, Bb);
    zero_f<<<cdivu(Bb, 256), 256, 0, stream>>>(sbuf, Bb);
    zero_f<<<cdivu(Bb * 32, 256), 256, 0, stream>>>(rbuf, Bb * 32);
    s2s_logit_kernel<<<cdivu(K, 256), 256, 0, stream>>>(xv, hbuf, seg, logit, mord, K);
    s2s_exp_kernel<<<cdivu(K, 256), 256, 0, stream>>>(logit, seg, mord, exbuf, sbuf, K);
    s2s_r_kernel<<<cdivu((long)K * 32, 256), 256, 0, stream>>>(xv, seg, exbuf, sbuf, rbuf, (long)K * 32);
    s2s_qstar_kernel<<<cdivu(Bb * 32, 256), 256, 0, stream>>>(hbuf, rbuf, qstar, Bb * 32);
  }
}

extern "C" void kernel_launch(void* const* d_in, const int* in_sizes, int n_in,
                              void* d_out, int out_size, void* d_ws, size_t ws_size,
                              hipStream_t stream) {
  (void)in_sizes; (void)n_in; (void)out_size; (void)ws_size;

  const int*   x     = (const int*)  d_in[0];
  const float* eattr = (const float*)d_in[1];
  const float* state = (const float*)d_in[2];
  const int*   eidx  = (const int*)  d_in[3];
  const int*   batch = (const int*)  d_in[4];
  const int* row = eidx;
  const int* col = eidx + Ee;

  // params leaves: blocks (sorted mlp keys, each W1,W2,b1,b2), dense1, dense2,
  // embedding, out, s2s_e, s2s_v
  const int P = 5;
  auto blkp = [&](int i, int mlp, int t) -> const float* {
    return (const float*)d_in[P + i * 24 + mlp * 4 + t];
  };
  enum { E_DENSE = 0, EDGE_MLP = 1, NODE_MLP = 2, STATE_MLP = 3, U_DENSE = 4, V_DENSE = 5 };
  const int T = P + 72;
  const float* d1W  = (const float*)d_in[T + 0];
  const float* d1b  = (const float*)d_in[T + 1];
  const float* d2W  = (const float*)d_in[T + 2];
  const float* d2b  = (const float*)d_in[T + 3];
  const float* emb  = (const float*)d_in[T + 4];
  const float* oW   = (const float*)d_in[T + 5];
  const float* ob   = (const float*)d_in[T + 6];
  S2SParams s2se{(const float*)d_in[T + 8], (const float*)d_in[T + 7],
                 (const float*)d_in[T + 10], (const float*)d_in[T + 9]};   // Wih, Whh, bih, bhh
  S2SParams s2sv{(const float*)d_in[T + 12], (const float*)d_in[T + 11],
                 (const float*)d_in[T + 14], (const float*)d_in[T + 13]};

  // workspace bump allocator
  char* wp = (char*)d_ws;
  auto alloc = [&](size_t bytes) -> char* {
    char* r = wp;
    wp += (bytes + 255) & ~(size_t)255;
    return r;
  };
  float* e_state = (float*)alloc((size_t)Ee * 32 * 4);
  float* e_d     = (float*)alloc((size_t)Ee * 32 * 4);
  float* e_out   = (float*)alloc((size_t)Ee * 32 * 4);
  float* v_state = (float*)alloc((size_t)Nn * 32 * 4);
  float* v_d     = (float*)alloc((size_t)Nn * 32 * 4);
  float* v_out   = (float*)alloc((size_t)Nn * 32 * 4);
  float* agg     = (float*)alloc((size_t)Nn * 32 * 4);
  float* cnt_n   = (float*)alloc((size_t)Nn * 4);
  float* u_state = (float*)alloc((size_t)Bb * 32 * 4);
  float* u_d     = (float*)alloc((size_t)Bb * 32 * 4);
  float* u_out   = (float*)alloc((size_t)Bb * 32 * 4);
  float* mean_v  = (float*)alloc((size_t)Bb * 32 * 4);
  float* mean_e  = (float*)alloc((size_t)Bb * 32 * 4);
  float* cntB    = (float*)alloc((size_t)Bb * 4);
  int*   ebatch  = (int*)  alloc((size_t)Ee * 4);
  float* logit   = (float*)alloc((size_t)Ee * 4);
  float* exbuf   = (float*)alloc((size_t)Ee * 4);
  unsigned* mord = (unsigned*)alloc((size_t)Bb * 4);
  float* sbuf    = (float*)alloc((size_t)Bb * 4);
  float* rbuf    = (float*)alloc((size_t)Bb * 32 * 4);
  float* hbuf    = (float*)alloc((size_t)Bb * 32 * 4);
  float* cbuf    = (float*)alloc((size_t)Bb * 32 * 4);
  float* qstar_v = (float*)alloc((size_t)Bb * 64 * 4);
  float* qstar_e = (float*)alloc((size_t)Bb * 64 * 4);

  // init
  embed_k<<<cdivu((long)Nn * 32, 256), 256, 0, stream>>>(x, emb, v_state, (long)Nn * 32);
  copy_f<<<cdivu(Bb * 32, 256), 256, 0, stream>>>(u_state, state, Bb * 32);
  ebatch_k<<<cdivu(Ee, 256), 256, 0, stream>>>(row, batch, ebatch, Ee);

  for (int i = 0; i < 3; ++i) {
    // e_dense
    if (i == 0)
      mlp_wmma_kernel<100, 128, 1><<<cdivu(Ee, 64), 128, 0, stream>>>(
          nullptr, eattr, nullptr, nullptr, nullptr, nullptr, nullptr, nullptr, nullptr,
          blkp(i, E_DENSE, 0), blkp(i, E_DENSE, 2), blkp(i, E_DENSE, 1), blkp(i, E_DENSE, 3),
          e_d, Ee);
    else
      mlp_wmma_kernel<32, 32, 0><<<cdivu(Ee, 64), 128, 0, stream>>>(
          e_state, nullptr, nullptr, nullptr, nullptr, nullptr, nullptr, nullptr, nullptr,
          blkp(i, E_DENSE, 0), blkp(i, E_DENSE, 2), blkp(i, E_DENSE, 1), blkp(i, E_DENSE, 3),
          e_d, Ee);
    // v_dense, u_dense
    mlp_wmma_kernel<32, 32, 0><<<cdivu(Nn, 64), 128, 0, stream>>>(
        v_state, nullptr, nullptr, nullptr, nullptr, nullptr, nullptr, nullptr, nullptr,
        blkp(i, V_DENSE, 0), blkp(i, V_DENSE, 2), blkp(i, V_DENSE, 1), blkp(i, V_DENSE, 3),
        v_d, Nn);
    mlp_wmma_kernel<32, 32, 0><<<cdivu(Bb, 64), 128, 0, stream>>>(
        u_state, nullptr, nullptr, nullptr, nullptr, nullptr, nullptr, nullptr, nullptr,
        blkp(i, U_DENSE, 0), blkp(i, U_DENSE, 2), blkp(i, U_DENSE, 1), blkp(i, U_DENSE, 3),
        u_d, Bb);
    // edge update
    mlp_wmma_kernel<128, 128, 2><<<cdivu(Ee, 64), 128, 0, stream>>>(
        nullptr, nullptr, row, col, batch, v_d, e_d, u_d, nullptr,
        blkp(i, EDGE_MLP, 0), blkp(i, EDGE_MLP, 2), blkp(i, EDGE_MLP, 1), blkp(i, EDGE_MLP, 3),
        e_out, Ee);
    // agg = seg_mean(e_out, col, N)
    zero_f<<<cdivu((long)Nn * 32, 256), 256, 0, stream>>>(agg, (long)Nn * 32);
    zero_f<<<cdivu(Nn, 256), 256, 0, stream>>>(cnt_n, Nn);
    scatter_mean_accum<<<cdivu((long)Ee * 32, 256), 256, 0, stream>>>(e_out, col, agg, cnt_n, (long)Ee * 32);
    div_cnt<<<cdivu((long)Nn * 32, 256), 256, 0, stream>>>(agg, cnt_n, (long)Nn * 32);
    // node update
    mlp_wmma_kernel<96, 96, 3><<<cdivu(Nn, 64), 128, 0, stream>>>(
        nullptr, nullptr, nullptr, nullptr, batch, v_d, nullptr, u_d, agg,
        blkp(i, NODE_MLP, 0), blkp(i, NODE_MLP, 2), blkp(i, NODE_MLP, 1), blkp(i, NODE_MLP, 3),
        v_out, Nn);
    // graph means
    zero_f<<<cdivu(Bb * 32, 256), 256, 0, stream>>>(mean_v, Bb * 32);
    zero_f<<<cdivu(Bb, 256), 256, 0, stream>>>(cntB, Bb);
    scatter_mean_accum<<<cdivu((long)Nn * 32, 256), 256, 0, stream>>>(v_out, batch, mean_v, cntB, (long)Nn * 32);
    div_cnt<<<cdivu(Bb * 32, 256), 256, 0, stream>>>(mean_v, cntB, Bb * 32);
    zero_f<<<cdivu(Bb * 32, 256), 256, 0, stream>>>(mean_e, Bb * 32);
    zero_f<<<cdivu(Bb, 256), 256, 0, stream>>>(cntB, Bb);
    scatter_mean_accum<<<cdivu((long)Ee * 32, 256), 256, 0, stream>>>(e_out, ebatch, mean_e, cntB, (long)Ee * 32);
    div_cnt<<<cdivu(Bb * 32, 256), 256, 0, stream>>>(mean_e, cntB, Bb * 32);
    // state update
    mlp_wmma_kernel<96, 96, 3><<<cdivu(Bb, 64), 128, 0, stream>>>(
        nullptr, nullptr, nullptr, nullptr, nullptr, mean_v, nullptr, u_d, mean_e,
        blkp(i, STATE_MLP, 0), blkp(i, STATE_MLP, 2), blkp(i, STATE_MLP, 1), blkp(i, STATE_MLP, 3),
        u_out, Bb);
    // residuals
    const float* e_res = (i == 0) ? e_d : e_state;
    const float* v_res = (i == 0) ? v_d : v_state;
    const float* u_res = (i == 0) ? u_d : u_state;
    add_f<<<cdivu((long)Ee * 32, 256), 256, 0, stream>>>(e_state, e_out, e_res, (long)Ee * 32);
    add_f<<<cdivu((long)Nn * 32, 256), 256, 0, stream>>>(v_state, v_out, v_res, (long)Nn * 32);
    add_f<<<cdivu(Bb * 32, 256), 256, 0, stream>>>(u_state, u_out, u_res, Bb * 32);
  }

  // Set2Set readouts
  run_set2set(v_state, batch, Nn, s2sv, hbuf, cbuf, qstar_v, logit, exbuf, mord, sbuf, rbuf, stream);
  run_set2set(e_state, ebatch, Ee, s2se, hbuf, cbuf, qstar_e, logit, exbuf, mord, sbuf, rbuf, stream);

  // head
  head_kernel<<<cdivu(Bb, 256), 256, 0, stream>>>(qstar_v, qstar_e, u_state,
                                                  d1W, d1b, d2W, d2b, oW, ob,
                                                  (float*)d_out, Bb);
}
